// MSAColAttention_58918361366844
// MI455X (gfx1250) — compile-verified, matrix operand order checked by reading, not run on hardware
//
#include <hip/hip_runtime.h>
#include <math.h>

// MSA column attention for MI455X (gfx1250, wave32, WMMA f16 16x16x32).
// Flash-style attention avoids the 549MB logits tensor; f16 data / f32 accum.
// K/V tiles staged into LDS via the Tensor Data Mover (tensor_load_to_lds).
// Workspace layout (needs ~160.3 MiB):
//   [0)          wT_qkvg  f16 [1024][128]   (qkv||gate weights, transposed)
//   [262144)     w_outT   f16 [128][256]
//   [327680)     Q        f16 [8][256][256][32]   ([h][l][seq][d])
//   +33554432    K        f16 same (pre-scaled by 1/sqrt(32))
//   +33554432    Vt       f16 [8][256][32][256]   ([h][l][d][seq], transposed)
//   +33554432    G        f16 [65536][256]  (sigmoid gate, [tok][h*32+d])
//   +33554432    O        f16 [65536][256]  (gated attention out)

typedef __attribute__((ext_vector_type(16))) _Float16 v16h;
typedef __attribute__((ext_vector_type(8)))  _Float16 v8h;
typedef __attribute__((ext_vector_type(8)))  float    v8f;
typedef __attribute__((ext_vector_type(4)))  unsigned int v4u;
typedef __attribute__((ext_vector_type(8)))  int      v8i;
typedef __attribute__((ext_vector_type(4)))  int      v4i;

__device__ __forceinline__ v16h cat16(v8h lo, v8h hi) {
  v16h r;
#pragma unroll
  for (int i = 0; i < 8; ++i) { r[i] = lo[i]; r[i + 8] = hi[i]; }
  return r;
}

__device__ __forceinline__ v8f wmma16(v16h a, v16h b, v8f c) {
  // emits v_wmma_f32_16x16x32_f16
  return __builtin_amdgcn_wmma_f32_16x16x32_f16(false, a, false, b, (short)0, c,
                                                false, false);
}

// TDM 1-D (degenerate 2-D) copy of `nelem` f16 elements global -> LDS.
// D# built per CDNA5 ISA §8.3/8.4: group0 = count|lds_addr|global_addr|type=2,
// group1 = data_size=2B, tensor_dim0 = tile_dim0 = nelem, tensor_dim1 = tile_dim1 = 1.
// This toolchain exposes the 6-arg builtin: (g0, g1, g2, g3, g4, cpol).
__device__ __forceinline__ void tdm_load_f16(unsigned int lds_addr,
                                             const _Float16* gptr,
                                             unsigned int nelem) {
  unsigned long long ga = (unsigned long long)(size_t)gptr;
  v4u g0;
  g0[0] = 1u;                                   // count=1, user mode
  g0[1] = lds_addr;                             // LDS byte address
  g0[2] = (unsigned int)ga;                     // global_addr[31:0]
  g0[3] = (unsigned int)((ga >> 32) & 0x01FFFFFFu) | (2u << 30);  // [56:32], type=2
  v8i g1;
  g1[0] = (int)(1u << 16);                      // data_size=1 (2 bytes), mask=0
  g1[1] = (int)((nelem & 0xFFFFu) << 16);       // tensor_dim0[15:0]
  g1[2] = (int)(((nelem >> 16) & 0xFFFFu) | (1u << 16)); // td0 hi | tensor_dim1=1
  g1[3] = (int)((nelem & 0xFFFFu) << 16);       // td1 hi(0) | tile_dim0 = nelem
  g1[4] = 1;                                    // tile_dim1=1, tile_dim2=0
  g1[5] = (int)nelem;                           // tensor_dim0_stride lo
  g1[6] = 0;                                    // td0_stride hi | td1_stride lo
  g1[7] = 0;
  v4i z4 = {0, 0, 0, 0};
  v8i z8 = {0, 0, 0, 0, 0, 0, 0, 0};
  __builtin_amdgcn_tensor_load_to_lds(g0, g1, z4, z4, z8, 0);
}

// ---------------------------------------------------------------- kernel 0
__global__ __launch_bounds__(256) void k_convert(
    const float* __restrict__ w_qkv, const float* __restrict__ w_g,
    const float* __restrict__ w_out, _Float16* __restrict__ wT,
    _Float16* __restrict__ w_outT) {
  int idx = blockIdx.x * 256 + threadIdx.x;
  if (idx < 1024 * 128) {
    int n = idx >> 7, k = idx & 127;
    float v = (n < 768) ? w_qkv[k * 768 + n] : w_g[k * 256 + (n - 768)];
    wT[idx] = (_Float16)v;
  } else {
    int i2 = idx - 1024 * 128;
    if (i2 < 128 * 256) {
      int n = i2 >> 8, k = i2 & 255;
      w_outT[i2] = (_Float16)w_out[k * 128 + n];
    }
  }
}

// ---------------------------------------------------------------- kernel 1
// Block = 128 threads (4 waves), 16 tokens per block. LN then WMMA projection
// D = W^T x mn^T  (M = out-col, N = token), 64 col-tiles split over 4 waves.
__global__ __launch_bounds__(128) void k_ln_proj(
    const float* __restrict__ m, const float* __restrict__ ln_g,
    const float* __restrict__ ln_b, const _Float16* __restrict__ wT,
    const float* __restrict__ b_g, _Float16* __restrict__ Q,
    _Float16* __restrict__ K, _Float16* __restrict__ Vt,
    _Float16* __restrict__ G) {
  __shared__ __align__(16) _Float16 s_mn[16 * 128];
  __shared__ float red_s[16][8], red_q[16][8];
  __shared__ float s_mu[16], s_rs[16];

  const int tid = threadIdx.x;
  const int tok = tid >> 3;   // 0..15
  const int part = tid & 7;   // 0..7 (16 features each)
  const int tok_g0 = blockIdx.x * 16;

  // ---- LayerNorm over 128 features ----
  float x[16];
  {
    const float* mp = m + (size_t)(tok_g0 + tok) * 128 + part * 16;
    float s = 0.f, q2 = 0.f;
#pragma unroll
    for (int i = 0; i < 16; ++i) { x[i] = mp[i]; s += x[i]; q2 += x[i] * x[i]; }
    red_s[tok][part] = s; red_q[tok][part] = q2;
  }
  __syncthreads();
  if (part == 0) {
    float ts = 0.f, tq = 0.f;
#pragma unroll
    for (int i = 0; i < 8; ++i) { ts += red_s[tok][i]; tq += red_q[tok][i]; }
    float mu = ts * (1.0f / 128.0f);
    float var = tq * (1.0f / 128.0f) - mu * mu;
    s_mu[tok] = mu;
    s_rs[tok] = rsqrtf(var + 1e-5f);
  }
  __syncthreads();
  {
    float mu = s_mu[tok], rs = s_rs[tok];
#pragma unroll
    for (int i = 0; i < 16; ++i) {
      int f = part * 16 + i;
      float y = (x[i] - mu) * rs * ln_g[f] + ln_b[f];
      s_mn[tok * 128 + f] = (_Float16)y;
    }
  }
  __syncthreads();

  // ---- Projection via WMMA ----
  const int lane = tid & 31, wave = tid >> 5;
  const int li = lane & 15, hi = lane >> 4;
  const int tg = tok_g0 + li;        // this lane's token (as N column)
  const int seq = tg >> 8, lcol = tg & 255;

  // B operand: mn^T, column = token li, K-dim = feature
  v16h bmat[4];
#pragma unroll
  for (int ks = 0; ks < 4; ++ks) {
    const _Float16* bp = &s_mn[li * 128 + ks * 32 + hi * 8];
    bmat[ks] = cat16(*(const v8h*)bp, *(const v8h*)(bp + 16));
  }

  for (int ct = wave; ct < 64; ct += 4) {
    // preload all A tiles so loads clause and overlap the WMMA chain
    v16h a[4];
#pragma unroll
    for (int ks = 0; ks < 4; ++ks) {
      const _Float16* ap = wT + (ct * 16 + li) * 128 + ks * 32 + hi * 8;
      a[ks] = cat16(*(const v8h*)ap, *(const v8h*)(ap + 16));
    }
    v8f acc = {};
#pragma unroll
    for (int ks = 0; ks < 4; ++ks) acc = wmma16(a[ks], bmat[ks], acc);

    // acc VGPR r -> out-col c = ct*16 + r + hi*8, token column = tg
    if (ct < 48) {                       // q | k | v
      int c0 = ct * 16;
      int sec = c0 >> 8;                 // 0=q 1=k 2=v
      int h = (c0 & 255) >> 5;
      int d0 = (c0 & 31) + hi * 8;
      if (sec == 0) {
        v8h o;
#pragma unroll
        for (int r = 0; r < 8; ++r) o[r] = (_Float16)acc[r];
        *(v8h*)(Q + ((h * 256 + lcol) * 256 + seq) * 32 + d0) = o;
      } else if (sec == 1) {             // fold 1/sqrt(32) into K
        v8h o;
#pragma unroll
        for (int r = 0; r < 8; ++r)
          o[r] = (_Float16)(acc[r] * 0.17677669529663687f);
        *(v8h*)(K + ((h * 256 + lcol) * 256 + seq) * 32 + d0) = o;
      } else {                           // V stored transposed: [h][l][d][seq]
        size_t vb = ((size_t)(h * 256 + lcol) * 32);
#pragma unroll
        for (int r = 0; r < 8; ++r)
          Vt[(vb + d0 + r) * 256 + seq] = (_Float16)acc[r];
      }
    } else {                             // gate
      int cg0 = (ct - 48) * 16 + hi * 8;
      v8h o;
#pragma unroll
      for (int r = 0; r < 8; ++r) {
        float z = acc[r] + b_g[cg0 + r];
        o[r] = (_Float16)__builtin_amdgcn_rcpf(1.0f + __expf(-z));
      }
      *(v8h*)(G + (size_t)tg * 256 + cg0) = o;
    }
  }
}

// ---------------------------------------------------------------- kernel 2
// One block per (l, h). Flash attention over MSA rows; S^T = K Q^T so softmax
// reductions are per-lane + one shfl_xor(16), and the softmaxed tile is
// already in B-operand layout for O^T = V^T P (no transpose shuffles).
// K and V^T slabs (16KB each) staged into LDS by the Tensor Data Mover.
__global__ __launch_bounds__(128) void k_attn(
    const _Float16* __restrict__ Q, const _Float16* __restrict__ K,
    const _Float16* __restrict__ Vt, const _Float16* __restrict__ G,
    _Float16* __restrict__ O) {
  __shared__ __align__(16) _Float16 sK[256 * 32];   // [j][d]
  __shared__ __align__(16) _Float16 sVt[32 * 256];  // [d][j]
  const int l = blockIdx.x;
  const int h = blockIdx.y;
  const int tid = threadIdx.x;
  const int lane = tid & 31, wave = tid >> 5;
  const size_t slab = (size_t)(h * 256 + l) * 8192;  // elements per (h,l) slab

  if (wave == 0) {  // one wave issues both DMA descriptors
    tdm_load_f16((unsigned int)(size_t)(void*)sK, K + slab, 8192u);
    tdm_load_f16((unsigned int)(size_t)(void*)sVt, Vt + slab, 8192u);
    __builtin_amdgcn_s_wait_tensorcnt(0);
  }
  __syncthreads();

  const int li = lane & 15, hi = lane >> 4;

  for (int it = wave; it < 16; it += 4) {
    const int iq = it * 16 + li;  // this lane's query row (N column of S^T)
    const _Float16* qp = Q + slab + iq * 32 + hi * 8;
    v16h qb = cat16(*(const v8h*)qp, *(const v8h*)(qp + 16));

    v8f acc0 = {}, acc1 = {};
    float m_run = -1e30f, l_run = 0.f;

    for (int js = 0; js < 8; ++js) {
      const int jb = js * 32;
      const _Float16* kp0 = &sK[(jb + li) * 32 + hi * 8];
      const _Float16* kp1 = &sK[(jb + 16 + li) * 32 + hi * 8];
      v16h ka0 = cat16(*(const v8h*)kp0, *(const v8h*)(kp0 + 16));
      v16h ka1 = cat16(*(const v8h*)kp1, *(const v8h*)(kp1 + 16));
      v8f z = {};
      v8f s0 = wmma16(ka0, qb, z);   // S^T rows j=jb..jb+15
      v8f s1 = wmma16(ka1, qb, z);   // S^T rows j=jb+16..jb+31

      float tm = s0[0];
#pragma unroll
      for (int r = 1; r < 8; ++r) tm = fmaxf(tm, s0[r]);
#pragma unroll
      for (int r = 0; r < 8; ++r) tm = fmaxf(tm, s1[r]);
      tm = fmaxf(tm, __shfl_xor(tm, 16, 32));   // combine lane-half partners
      float m_new = fmaxf(m_run, tm);
      float corr = __expf(m_run - m_new);

      float rsum = 0.f;
      v16h pb;
#pragma unroll
      for (int r = 0; r < 8; ++r) {
        float p0 = __expf(s0[r] - m_new);
        float p1 = __expf(s1[r] - m_new);
        rsum += p0 + p1;
        pb[r] = (_Float16)p0;       // already in B-operand element order
        pb[r + 8] = (_Float16)p1;
      }
      rsum += __shfl_xor(rsum, 16, 32);
      l_run = l_run * corr + rsum;
      m_run = m_new;
#pragma unroll
      for (int r = 0; r < 8; ++r) { acc0[r] *= corr; acc1[r] *= corr; }

      const _Float16* vp0 = &sVt[li * 256 + jb + hi * 8];          // d 0..15
      const _Float16* vp1 = &sVt[(16 + li) * 256 + jb + hi * 8];   // d 16..31
      v16h va0 = cat16(*(const v8h*)vp0, *(const v8h*)(vp0 + 16));
      v16h va1 = cat16(*(const v8h*)vp1, *(const v8h*)(vp1 + 16));
      acc0 = wmma16(va0, pb, acc0);  // O^T rows d=0..15
      acc1 = wmma16(va1, pb, acc1);  // O^T rows d=16..31
    }

    float inv = __builtin_amdgcn_rcpf(l_run);
    const size_t ob = (((size_t)iq * 256 + l) * 256 + h * 32);
    v8h g0 = *(const v8h*)(G + ob + hi * 8);
    v8h g1 = *(const v8h*)(G + ob + 16 + hi * 8);
    v8h o0, o1;
#pragma unroll
    for (int r = 0; r < 8; ++r) {
      o0[r] = (_Float16)(acc0[r] * inv * (float)g0[r]);
      o1[r] = (_Float16)(acc1[r] * inv * (float)g1[r]);
    }
    *(v8h*)(O + ob + hi * 8) = o0;
    *(v8h*)(O + ob + 16 + hi * 8) = o1;
  }
}

// ---------------------------------------------------------------- kernel 3
// out = O_g @ w_out + b_out ; D = w_outT x O^T (M=n, N=token) so each lane
// stores 8 contiguous f32 outputs.
__global__ __launch_bounds__(128) void k_out(
    const _Float16* __restrict__ O, const _Float16* __restrict__ w_outT,
    const float* __restrict__ b_out, float* __restrict__ out) {
  const int tid = threadIdx.x;
  const int lane = tid & 31, wave = tid >> 5;
  const int li = lane & 15, hi = lane >> 4;
  const int tok = blockIdx.x * 16 + li;

  // preload this lane's B operands (O rows for its token)
  v16h bmat[8];
#pragma unroll
  for (int ks = 0; ks < 8; ++ks) {
    const _Float16* bp = O + (size_t)tok * 256 + ks * 32 + hi * 8;
    bmat[ks] = cat16(*(const v8h*)bp, *(const v8h*)(bp + 16));
  }

  for (int nt = wave; nt < 8; nt += 4) {
    v8f acc = {};
#pragma unroll
    for (int kh = 0; kh < 2; ++kh) {
      v16h a[4];
#pragma unroll
      for (int k4 = 0; k4 < 4; ++k4) {
        int ks = kh * 4 + k4;
        const _Float16* ap = w_outT + (nt * 16 + li) * 256 + ks * 32 + hi * 8;
        a[k4] = cat16(*(const v8h*)ap, *(const v8h*)(ap + 16));
      }
#pragma unroll
      for (int k4 = 0; k4 < 4; ++k4) acc = wmma16(a[k4], bmat[kh * 4 + k4], acc);
    }
    int n0 = nt * 16 + hi * 8;
    float4 b0 = *(const float4*)(b_out + n0);
    float4 b1 = *(const float4*)(b_out + n0 + 4);
    float4 r0, r1;
    r0.x = acc[0] + b0.x; r0.y = acc[1] + b0.y;
    r0.z = acc[2] + b0.z; r0.w = acc[3] + b0.w;
    r1.x = acc[4] + b1.x; r1.y = acc[5] + b1.y;
    r1.z = acc[6] + b1.z; r1.w = acc[7] + b1.w;
    *(float4*)(out + (size_t)tok * 128 + n0) = r0;
    *(float4*)(out + (size_t)tok * 128 + n0 + 4) = r1;
  }
}

// ---------------------------------------------------------------- launch
extern "C" void kernel_launch(void* const* d_in, const int* in_sizes, int n_in,
                              void* d_out, int out_size, void* d_ws,
                              size_t ws_size, hipStream_t stream) {
  const float* m = (const float*)d_in[0];
  const float* ln_g = (const float*)d_in[1];
  const float* ln_b = (const float*)d_in[2];
  const float* w_qkv = (const float*)d_in[3];
  const float* w_g = (const float*)d_in[4];
  const float* b_g = (const float*)d_in[5];
  const float* w_out = (const float*)d_in[6];
  const float* b_out = (const float*)d_in[7];
  float* out = (float*)d_out;

  char* ws = (char*)d_ws;
  const size_t QKV_BYTES = (size_t)8 * 256 * 256 * 32 * 2;  // 33,554,432
  _Float16* wT = (_Float16*)(ws);
  _Float16* w_outT = (_Float16*)(ws + 262144);
  _Float16* Qb = (_Float16*)(ws + 327680);
  _Float16* Kb = (_Float16*)(ws + 327680 + QKV_BYTES);
  _Float16* Vtb = (_Float16*)(ws + 327680 + 2 * QKV_BYTES);
  _Float16* Gb = (_Float16*)(ws + 327680 + 3 * QKV_BYTES);
  _Float16* Ob = (_Float16*)(ws + 327680 + 4 * QKV_BYTES);

  k_convert<<<640, 256, 0, stream>>>(w_qkv, w_g, w_out, wT, w_outT);
  k_ln_proj<<<4096, 128, 0, stream>>>(m, ln_g, ln_b, wT, b_g, Qb, Kb, Vtb, Gb);
  k_attn<<<dim3(256, 8), 128, 0, stream>>>(Qb, Kb, Vtb, Gb, Ob);
  k_out<<<4096, 128, 0, stream>>>(Ob, w_outT, b_out, out);
}
